// CA3CA1Circuit_87935160418560
// MI455X (gfx1250) — matrix operand order repaired and checked by reading, not compiled, not run on hardware
//
#include <hip/hip_runtime.h>
#include <hip/hip_bf16.h>
#include <math.h>

typedef __attribute__((ext_vector_type(16))) _Float16 v16h;
typedef __attribute__((ext_vector_type(8)))  float    v8f;

#define N_CA3 2048
#define N_CA1 8192
#define DELAY 5
#define THETA_AMP 0.3f
#define PERF 0.2f
#define LTP_TH 0.7f
#define LTD_TH 0.3f
#define LR 0.001f

// ---------------------------------------------------------------------------
// Wave-level matvec: y[row_base .. row_base+15] += W[16 x K] @ x[K] using
// v_wmma_f32_16x16x32_f16. Weights/vector are f32 in memory, converted to f16
// for the matrix op, accumulated in f32 (C/D regs). The vector is replicated
// across all 16 B-columns so every column of D holds the same matvec result.
//
// A layout (16-bit A, 16x32, ISA 7.12.2): lanes 0-15 -> M=lane, K {0..7,16..23};
// lanes 16-31 -> M=lane-16, K {8..15,24..31}.
// B layout (16-bit B, 32x16): lanes 0-15 -> N=lane, K {0..7,16..23}; lanes 16-31 ->
// N=lane-16, K {8..15,24..31}.
// D layout: VGPR r: lanes 0-15 -> (M=r, N=lane); lanes 16-31 -> (M=8+r).
// ---------------------------------------------------------------------------
__device__ __forceinline__ v8f wmma_matvec16(const float* __restrict__ W,
                                             const float* __restrict__ x,
                                             int row_base, int K)
{
    const int lane = threadIdx.x & 31;
    const int lh   = lane >> 4;                 // which half-wave
    const int m    = row_base + (lane & 15);
    const float* Wr = W + (size_t)m * (size_t)K;

    v8f acc = {0.f, 0.f, 0.f, 0.f, 0.f, 0.f, 0.f, 0.f};

    for (int k0 = 0; k0 < K; k0 += 32) {
        const int ka = k0 + (lh ? 8 : 0);       // A: first 8-wide K chunk
        const int kb = k0 + (lh ? 8 : 0);       // B: first 8-wide K chunk
        const float4* pa0 = (const float4*)(Wr + ka);
        const float4* pa1 = (const float4*)(Wr + ka + 16);
        float4 a0 = pa0[0], a1 = pa0[1];
        float4 a2 = pa1[0], a3 = pa1[1];
        const float4* pb = (const float4*)(x + kb);     // broadcast within half-wave
        float4 b0 = pb[0], b1 = pb[1], b2 = pb[4], b3 = pb[5];

        v16h A, B;
        A[0]=(_Float16)a0.x;  A[1]=(_Float16)a0.y;  A[2]=(_Float16)a0.z;  A[3]=(_Float16)a0.w;
        A[4]=(_Float16)a1.x;  A[5]=(_Float16)a1.y;  A[6]=(_Float16)a1.z;  A[7]=(_Float16)a1.w;
        A[8]=(_Float16)a2.x;  A[9]=(_Float16)a2.y;  A[10]=(_Float16)a2.z; A[11]=(_Float16)a2.w;
        A[12]=(_Float16)a3.x; A[13]=(_Float16)a3.y; A[14]=(_Float16)a3.z; A[15]=(_Float16)a3.w;
        B[0]=(_Float16)b0.x;  B[1]=(_Float16)b0.y;  B[2]=(_Float16)b0.z;  B[3]=(_Float16)b0.w;
        B[4]=(_Float16)b1.x;  B[5]=(_Float16)b1.y;  B[6]=(_Float16)b1.z;  B[7]=(_Float16)b1.w;
        B[8]=(_Float16)b2.x;  B[9]=(_Float16)b2.y;  B[10]=(_Float16)b2.z; B[11]=(_Float16)b2.w;
        B[12]=(_Float16)b3.x; B[13]=(_Float16)b3.y; B[14]=(_Float16)b3.z; B[15]=(_Float16)b3.w;

        acc = __builtin_amdgcn_wmma_f32_16x16x32_f16(
            /*neg_a=*/false, A, /*neg_b=*/false, B,
            /*c_mod=*/(short)0, acc, /*reuse_a=*/false, /*reuse_b=*/false);
    }
    return acc;
}

// ---------------------------------------------------------------------------
__global__ void init_ws_kernel(float* __restrict__ ws)
{
    if (threadIdx.x < 3) ws[threadIdx.x] = 0.0f;
}

// ws[2] += sum |ca3_new| ; out_ca3 = ca3_new
__global__ void ca3_update_kernel(const float* __restrict__ W,
                                  const float* __restrict__ ca3_act,
                                  const float* __restrict__ cortical,
                                  const float* __restrict__ theta,
                                  float* __restrict__ out_ca3,
                                  float* __restrict__ ws)
{
    const int wave = (blockIdx.x * blockDim.x + threadIdx.x) >> 5;
    const int row_base = wave * 16;

    v8f acc = wmma_matvec16(W, ca3_act, row_base, N_CA3);

    const int lane = threadIdx.x & 31;
    if ((lane & 15) == 0) {
        const int lh = lane >> 4;
        // theta in [0, 2*pi): fast hw sin (v_sin_f32), no Payne-Hanek needed
        const float mod = 1.0f + THETA_AMP * __sinf(theta[0]);
        float local = 0.0f;
        #pragma unroll
        for (int r = 0; r < 8; ++r) {
            const int m = row_base + lh * 8 + r;
            const float total = (cortical[m] + acc[r]) * mod;
            float nv = ca3_act[m] * (1.0f - 1.0f / 20.0f) + tanhf(total) * (1.0f / 20.0f);
            nv = fminf(fmaxf(nv, 0.0f), 2.0f);
            out_ca3[m] = nv;
            local += fabsf(nv);
        }
        atomicAdd(&ws[2], local);
    }
}

// out_ca1 = CA1 update with delayed Schaffer input
__global__ void ca1_update_kernel(const float* __restrict__ Wsc,
                                  const float* __restrict__ ca1_act,
                                  const float* __restrict__ cortical,
                                  const float* __restrict__ theta,
                                  const float* __restrict__ delay_buf,
                                  const int* __restrict__ delay_pos,
                                  float* __restrict__ out_ca1)
{
    const int wave = (blockIdx.x * blockDim.x + threadIdx.x) >> 5;
    const int row_base = wave * 16;

    const int pos = delay_pos[0] % DELAY;
    const float* delayed = delay_buf + ((pos + 2) % DELAY) * N_CA3;

    v8f acc = wmma_matvec16(Wsc, delayed, row_base, N_CA3);

    const int lane = threadIdx.x & 31;
    if ((lane & 15) == 0) {
        const int lh = lane >> 4;
        const float mod = 1.0f + THETA_AMP * __sinf(theta[0] + 0.78539816339744831f);
        #pragma unroll
        for (int r = 0; r < 8; ++r) {
            const int m = row_base + lh * 8 + r;
            const float perf = (m < N_CA3) ? cortical[m] * PERF : 0.0f;
            const float total = (acc[r] + perf) * mod;
            float nv = ca1_act[m] * (1.0f - 1.0f / 15.0f) + tanhf(total) * (1.0f / 15.0f);
            nv = fminf(fmaxf(nv, 0.0f), 2.0f);
            out_ca1[m] = nv;
        }
    }
}

// ws[1] += sum |W_ca3 @ ca3_new|
__global__ void rec_strength_kernel(const float* __restrict__ W,
                                    const float* __restrict__ ca3_new,
                                    float* __restrict__ ws)
{
    const int wave = (blockIdx.x * blockDim.x + threadIdx.x) >> 5;
    const int row_base = wave * 16;

    v8f acc = wmma_matvec16(W, ca3_new, row_base, N_CA3);

    const int lane = threadIdx.x & 31;
    if ((lane & 15) == 0) {
        float local = 0.0f;
        #pragma unroll
        for (int r = 0; r < 8; ++r) local += fabsf(acc[r]);
        atomicAdd(&ws[1], local);
    }
}

__device__ __forceinline__ float plast_abs(float a, float d, float w, float m)
{
    const float corr = a * d;
    float ch = 0.0f;
    if (corr > LTP_TH)                      ch = (corr - LTP_TH) * m;
    else if (corr > 0.0f && corr < LTD_TH)  ch = -(LTD_TH - corr) * m * 0.3f;
    float wn = w + ch * LR;
    wn = fminf(fmaxf(wn, -1.0f), 1.0f);
    return fabsf(wn);
}

// ws[0] += sum |clip(W_schaffer + dW, -1, 1)| over the full 8192x2048 matrix
__global__ void plasticity_kernel(const float* __restrict__ Wsc,
                                  const float* __restrict__ meta,
                                  const float* __restrict__ ca1_new,
                                  const float* __restrict__ delay_buf,
                                  const int* __restrict__ delay_pos,
                                  float* __restrict__ ws)
{
    const int pos = delay_pos[0] % DELAY;
    const float* delayed = delay_buf + ((pos + 2) % DELAY) * N_CA3;

    const size_t total4 = (size_t)N_CA1 * (size_t)N_CA3 / 4;
    const size_t stride = (size_t)gridDim.x * blockDim.x;
    float local = 0.0f;

    for (size_t idx = (size_t)blockIdx.x * blockDim.x + threadIdx.x;
         idx < total4; idx += stride) {
        const size_t e = idx * 4;                 // 4 consecutive j in one row i
        const int i = (int)(e >> 11);             // / N_CA3
        const int j = (int)(e & (N_CA3 - 1));
        const float a  = ca1_new[i];
        const float4 d = *(const float4*)(delayed + j);
        const float4 w = *(const float4*)(Wsc + e);
        const float4 m = *(const float4*)(meta + e);
        local += plast_abs(a, d.x, w.x, m.x);
        local += plast_abs(a, d.y, w.y, m.y);
        local += plast_abs(a, d.z, w.z, m.z);
        local += plast_abs(a, d.w, w.w, m.w);
    }
    // wave32 reduction, one atomic per wave
    for (int off = 16; off > 0; off >>= 1)
        local += __shfl_down(local, off, 32);
    if ((threadIdx.x & 31) == 0) atomicAdd(&ws[0], local);
}

__global__ void copy_delayed_kernel(const float* __restrict__ delay_buf,
                                    const int* __restrict__ delay_pos,
                                    float* __restrict__ out_del)
{
    const int i = blockIdx.x * blockDim.x + threadIdx.x;
    if (i < N_CA3) {
        const int pos = delay_pos[0] % DELAY;
        out_del[i] = delay_buf[((pos + 2) % DELAY) * N_CA3 + i];
    }
}

__global__ void finalize_kernel(const float* __restrict__ ws,
                                float* __restrict__ out_sc)
{
    const float syn = ws[0] / (float)((size_t)N_CA1 * (size_t)N_CA3);
    const float rec = ws[1] / (float)N_CA3;
    const float tot = ws[2] / (float)N_CA3 + 1e-6f;
    float pc = rec / tot;
    pc = fminf(fmaxf(pc, 0.0f), 1.0f);
    out_sc[0] = syn;
    out_sc[1] = pc;
}

// ---------------------------------------------------------------------------
extern "C" void kernel_launch(void* const* d_in, const int* in_sizes, int n_in,
                              void* d_out, int out_size, void* d_ws, size_t ws_size,
                              hipStream_t stream)
{
    const float* cortical = (const float*)d_in[0];
    const float* theta    = (const float*)d_in[1];
    const float* w_ca3    = (const float*)d_in[2];
    const float* ca3_act  = (const float*)d_in[3];
    const float* ca1_act  = (const float*)d_in[4];
    const float* w_sc     = (const float*)d_in[5];
    const float* meta     = (const float*)d_in[6];
    // d_in[7] synaptic_tags: unused by the reference computation
    const float* dbuf     = (const float*)d_in[8];
    const int*   dpos     = (const int*)d_in[9];

    float* out     = (float*)d_out;
    float* out_ca3 = out;                          // [0, 2048)
    float* out_ca1 = out + N_CA3;                  // [2048, 10240)
    float* out_del = out + N_CA3 + N_CA1;          // [10240, 12288)
    float* out_sc  = out + N_CA3 + N_CA1 + N_CA3;  // [12288, 12290)
    float* ws      = (float*)d_ws;                 // 3 float accumulators

    init_ws_kernel<<<1, 32, 0, stream>>>(ws);
    // 2048 rows / 16 per wave = 128 waves = 16 blocks of 8 waves
    ca3_update_kernel<<<16, 256, 0, stream>>>(w_ca3, ca3_act, cortical, theta,
                                              out_ca3, ws);
    // 8192 rows / 16 per wave = 512 waves = 64 blocks
    ca1_update_kernel<<<64, 256, 0, stream>>>(w_sc, ca1_act, cortical, theta,
                                              dbuf, dpos, out_ca1);
    rec_strength_kernel<<<16, 256, 0, stream>>>(w_ca3, out_ca3, ws);
    plasticity_kernel<<<1024, 256, 0, stream>>>(w_sc, meta, out_ca1, dbuf, dpos, ws);
    copy_delayed_kernel<<<8, 256, 0, stream>>>(dbuf, dpos, out_del);
    finalize_kernel<<<1, 1, 0, stream>>>(ws, out_sc);
}
